// Attention_7739531067867
// MI455X (gfx1250) — matrix operand order repaired and checked by reading, not compiled, not run on hardware
//
#include <hip/hip_runtime.h>

// ---------------------------------------------------------------------------
// GQA causal attention forward for MI455X (gfx1250, wave32, WMMA bf16).
// Double-buffered LDS staging fed by GLOBAL_LOAD_ASYNC_TO_LDS_B128
// (ASYNCcnt); fragments preloaded so WMMAs issue in bursts behind one wait.
// ---------------------------------------------------------------------------

typedef __bf16 bf16;
typedef __attribute__((ext_vector_type(16))) __bf16 v16bf;
typedef __attribute__((ext_vector_type(8)))  __bf16 v8bf;
typedef __attribute__((ext_vector_type(8)))  float  v8f;
typedef int v4i __attribute__((vector_size(16)));

#define WMMA_BF16(a, b, c) \
  __builtin_amdgcn_wmma_f32_16x16x32_bf16(false, (a), false, (b), (short)0, (c), false, false)

#if __has_builtin(__builtin_amdgcn_global_load_async_to_lds_b128)
#define HAVE_ASYNC_LDS 1
#else
#define HAVE_ASYNC_LDS 0
#endif

// 16-byte global -> LDS copy: async DMA path on gfx1250, register fallback.
__device__ __forceinline__ void cp16(const bf16* g, bf16* l) {
#if HAVE_ASYNC_LDS
  __builtin_amdgcn_global_load_async_to_lds_b128((v4i*)g, (v4i*)l, 0, 0);
#else
  *(v8bf*)l = *(const v8bf*)g;
#endif
}

__device__ __forceinline__ void wait_async0() {
#if HAVE_ASYNC_LDS
#if __has_builtin(__builtin_amdgcn_s_wait_asynccnt)
  __builtin_amdgcn_s_wait_asynccnt(0);
#else
  asm volatile("s_wait_asynccnt 0" ::: "memory");
#endif
#endif
}

static constexpr int kB  = 2;
static constexpr int kT  = 2048;
static constexpr int kD  = 2048;
static constexpr int kH  = 16;
static constexpr int kKV = 4;
static constexpr int kHD = 128;
static constexpr int kRD = 64;
// 1/sqrt(128) * log2(e): softmax done in exp2 domain
static constexpr float kScaleLog2 = 0.08838834764831845f * 1.4426950408889634f;

// Load a 16x32 bf16 WMMA fragment as two 16-byte chunks.
__device__ __forceinline__ v16bf ld2(const bf16* p0, const bf16* p1) {
  union { v16bf v; v8bf h[2]; } u;
  u.h[0] = *(const v8bf*)p0;
  u.h[1] = *(const v8bf*)p1;
  return u.v;
}

// ---------------------------------------------------------------------------
// Elementwise prep kernels
// ---------------------------------------------------------------------------
__global__ void cvt_bf16_kernel(const float* __restrict__ in, bf16* __restrict__ out, int n) {
  int i = blockIdx.x * 256 + threadIdx.x;
  if (i < n) out[i] = (bf16)in[i];
}

// in: fp32 [R][C] row-major  ->  out: bf16 [C][R] row-major (i.e. transposed)
__global__ void transpose_cvt_kernel(const float* __restrict__ in, bf16* __restrict__ out,
                                     int R, int C) {
  int i = blockIdx.x * 256 + threadIdx.x;
  if (i < R * C) {
    int r = i / C, c = i % C;
    out[(size_t)c * R + r] = (bf16)in[i];
  }
}

// RoPE on Q: fp32 [B,T,H*HD] -> bf16 [B,H,T,HD]
__global__ void rope_q_kernel(const float* __restrict__ Qf, const float* __restrict__ cosT,
                              const float* __restrict__ sinT, bf16* __restrict__ Qb) {
  int i = blockIdx.x * 256 + threadIdx.x;          // B*T*H*(HD/2) = 2^22
  if (i >= kB * kT * kH * (kHD / 2)) return;
  int p = i & 63;
  int h = (i >> 6) & 15;
  int t = (i >> 10) & 2047;
  int b = i >> 21;
  const float* src = Qf + ((size_t)(b * kT + t)) * (kH * kHD) + h * kHD + 2 * p;
  float e = src[0], o = src[1];
  if (p < kRD / 2) {
    float c = cosT[t * (kRD / 2) + p], s = sinT[t * (kRD / 2) + p];
    float re = e * c - o * s;
    float ro = e * s + o * c;
    e = re; o = ro;
  }
  bf16* dst = Qb + ((size_t)(b * kH + h) * kT + t) * kHD + 2 * p;
  dst[0] = (bf16)e;
  dst[1] = (bf16)o;
}

// RoPE on K: fp32 [B,T,KV*HD] -> bf16 [B,KV,T,HD] + fp32 present_k [B,KV,T,HD]
__global__ void rope_k_kernel(const float* __restrict__ Kf, const float* __restrict__ cosT,
                              const float* __restrict__ sinT, bf16* __restrict__ Kb,
                              float* __restrict__ pk) {
  int i = blockIdx.x * 256 + threadIdx.x;          // B*T*KV*(HD/2) = 2^20
  if (i >= kB * kT * kKV * (kHD / 2)) return;
  int p  = i & 63;
  int kv = (i >> 6) & 3;
  int t  = (i >> 8) & 2047;
  int b  = i >> 19;
  const float* src = Kf + ((size_t)(b * kT + t)) * (kKV * kHD) + kv * kHD + 2 * p;
  float e = src[0], o = src[1];
  if (p < kRD / 2) {
    float c = cosT[t * (kRD / 2) + p], s = sinT[t * (kRD / 2) + p];
    float re = e * c - o * s;
    float ro = e * s + o * c;
    e = re; o = ro;
  }
  size_t di = ((size_t)(b * kKV + kv) * kT + t) * kHD + 2 * p;
  Kb[di] = (bf16)e; Kb[di + 1] = (bf16)o;
  pk[di] = e;       pk[di + 1] = o;
}

// V: fp32 [B,T,KV*HD] -> bf16 transposed [B,KV,HD,T] + fp32 present_v [B,KV,T,HD]
__global__ void v_prep_kernel(const float* __restrict__ Vf, bf16* __restrict__ VbT,
                              float* __restrict__ pv) {
  int i = blockIdx.x * 256 + threadIdx.x;          // B*T*KV*HD = 2^21
  if (i >= kB * kT * kKV * kHD) return;
  int c  = i & 127;
  int kv = (i >> 7) & 3;
  int t  = (i >> 9) & 2047;
  int b  = i >> 20;
  float v = Vf[((size_t)(b * kT + t)) * (kKV * kHD) + kv * kHD + c];
  pv[((size_t)(b * kKV + kv) * kT + t) * kHD + c] = v;
  VbT[((size_t)(b * kKV + kv) * kHD + c) * kT + t] = (bf16)v;
}

// ---------------------------------------------------------------------------
// bf16 WMMA GEMM, LDS double-buffered: C[M,N] = A[M,K] * B[K,N], B given
// transposed as BT[N,K]. 256 threads = 8 waves; block tile 128(M) x 128(N);
// 64-deep k-slices (128x64 bf16 = 16KB each) async-copied into LDS.
// Each wave: 32(M) x 64(N), 16 WMMAs per k-slice issued as two 8-WMMA bursts.
// ---------------------------------------------------------------------------
__global__ void gemm_bf16(const bf16* __restrict__ A, const bf16* __restrict__ BT,
                          float* __restrict__ C, int M, int N, int K) {
  __shared__ bf16 As[2][128][64];   // [m][k]
  __shared__ bf16 Bs[2][128][64];   // [n][k]

  const int tid   = threadIdx.x;
  const int lane  = tid & 31;
  const int wave  = tid >> 5;
  const int row16 = lane & 15;
  const int hi    = lane >> 4;
  const int wm    = wave & 3;       // 4 m-positions of 32 rows
  const int wn    = wave >> 2;      // 2 n-positions of 64 cols
  const int m0    = blockIdx.y * 128;
  const int n0    = blockIdx.x * 128;

  // staging: 1024 16B chunks per 128x64 tile, 4 per thread per tile
  const int srow0 = tid >> 1;             // paired with i gives rows 0..127
  const int schk0 = tid & 1;              // with i gives chunks 0..7

  v8f acc[2][4] = {};

  // ---- prologue: stage k=0 into buffer 0 ----
#pragma unroll
  for (int i = 0; i < 4; ++i) {
    int idx = tid + i * 256;
    int row = idx >> 3, chk = idx & 7;
    cp16(A  + (size_t)(m0 + row) * K + chk * 8, &As[0][row][chk * 8]);
    cp16(BT + (size_t)(n0 + row) * K + chk * 8, &Bs[0][row][chk * 8]);
  }

  int buf = 0;
  for (int k = 0; k < K; k += 64) {
    wait_async0();
    __syncthreads();
    if (k + 64 < K) {
      int nb = buf ^ 1;
      int kk = k + 64;
#pragma unroll
      for (int i = 0; i < 4; ++i) {
        int idx = tid + i * 256;
        int row = idx >> 3, chk = idx & 7;
        cp16(A  + (size_t)(m0 + row) * K + kk + chk * 8, &As[nb][row][chk * 8]);
        cp16(BT + (size_t)(n0 + row) * K + kk + chk * 8, &Bs[nb][row][chk * 8]);
      }
    }
    // ---- compute out of LDS: two 32-deep sub-steps ----
#pragma unroll
    for (int kk = 0; kk < 64; kk += 32) {
      v16bf af[2];
#pragma unroll
      for (int mt = 0; mt < 2; ++mt) {
        const bf16* ap = &As[buf][wm * 32 + mt * 16 + row16][kk];
        af[mt] = ld2(ap + hi * 8, ap + 16 + hi * 8);
      }
      v16bf bfr[4];
#pragma unroll
      for (int nt = 0; nt < 4; ++nt) {
        const bf16* bp = &Bs[buf][wn * 64 + nt * 16 + row16][kk + hi * 8];
        bfr[nt] = ld2(bp, bp + 16);
      }
#pragma unroll
      for (int nt = 0; nt < 4; ++nt)
#pragma unroll
        for (int mt = 0; mt < 2; ++mt)
          acc[mt][nt] = WMMA_BF16(af[mt], bfr[nt], acc[mt][nt]);
    }
    __syncthreads();
    buf ^= 1;
  }

#pragma unroll
  for (int mt = 0; mt < 2; ++mt) {
#pragma unroll
    for (int nt = 0; nt < 4; ++nt) {
#pragma unroll
      for (int r = 0; r < 8; ++r) {
        int mrow = m0 + wm * 32 + mt * 16 + r + hi * 8;
        int ncol = n0 + wn * 64 + nt * 16 + row16;
        C[(size_t)mrow * N + ncol] = acc[mt][nt][r];
      }
    }
  }
}

// ---------------------------------------------------------------------------
// Flash attention, causal, GQA (head h uses kv head h/4).
// One workgroup (4 waves) owns a 64-row q block of one head; each wave takes
// 16 q rows. K (32x128) and V (128x32) kv-tiles are async-staged into LDS
// (double buffered) and shared by all 4 waves. Fragments preloaded so each
// WMMA burst (8 score / 8 PV) sits behind a single dscnt wait.
// ---------------------------------------------------------------------------
__global__ void flash_kernel(const bf16* __restrict__ Qb, const bf16* __restrict__ Kb,
                             const bf16* __restrict__ VbT, bf16* __restrict__ Ob) {
  __shared__ bf16 Ks[2][32][128];   // [kv pos][channel]  8KB per buffer
  __shared__ bf16 Vs[2][128][32];   // [channel][kv pos]  8KB per buffer
  __shared__ bf16 plds[4][16][32];  // per-wave P staging (C-layout -> A-layout)

  const int tid   = threadIdx.x;
  const int lane  = tid & 31;
  const int wave  = tid >> 5;
  const int row16 = lane & 15;
  const int hi    = lane >> 4;

  const int bid = blockIdx.x;                 // B*H*(T/64) = 1024 blocks
  const int qblk = bid & 31;
  const int h    = (bid >> 5) & 15;
  const int b    = bid >> 9;
  const int kvh  = h >> 2;
  const int q0b  = qblk * 64;                 // workgroup q range [q0b, q0b+63]
  const int q0   = q0b + wave * 16;           // this wave's q rows

  const bf16* kbase = Kb  + (size_t)(b * kKV + kvh) * kT * kHD;
  const bf16* vbase = VbT + (size_t)(b * kKV + kvh) * kHD * kT;

  // Q fragments (persist across kv loop): 16 rows x 128 channels = 4 frags
  const bf16* qbase = Qb + ((size_t)(b * kH + h) * kT + q0 + row16) * kHD;
  v16bf qa[4];
#pragma unroll
  for (int s = 0; s < 4; ++s)
    qa[s] = ld2(qbase + 32 * s + hi * 8, qbase + 32 * s + 16 + hi * 8);

  v8f o[8] = {};                              // 16 x 128 output accum
  float mrow[8], lrow[8];
#pragma unroll
  for (int r = 0; r < 8; ++r) { mrow[r] = -1e30f; lrow[r] = 0.f; }

  const int jend = q0b + 63;                  // last kv row any wave needs

  // ---- prologue: stage j=0 tiles into buffer 0 ----
#pragma unroll
  for (int i = 0; i < 4; ++i) {
    int idx = tid + i * 128;
    int kr = idx >> 4, kc = idx & 15;         // K: 32 rows x 16 chunks
    cp16(kbase + (size_t)kr * kHD + kc * 8, &Ks[0][kr][kc * 8]);
    int vr = idx >> 2, vc = idx & 3;          // V: 128 rows x 4 chunks
    cp16(vbase + (size_t)vr * kT + vc * 8, &Vs[0][vr][vc * 8]);
  }

  int buf = 0;
  for (int j = 0; j <= jend; j += 32) {
    wait_async0();
    __syncthreads();
    if (j + 32 <= jend) {
      int nb = buf ^ 1;
      int jj = j + 32;
#pragma unroll
      for (int i = 0; i < 4; ++i) {
        int idx = tid + i * 128;
        int kr = idx >> 4, kc = idx & 15;
        cp16(kbase + (size_t)(jj + kr) * kHD + kc * 8, &Ks[nb][kr][kc * 8]);
        int vr = idx >> 2, vc = idx & 3;
        cp16(vbase + (size_t)vr * kT + jj + vc * 8, &Vs[nb][vr][vc * 8]);
      }
    }

    if (j <= q0 + 15) {                       // skip blocks fully above causal limit
      // ---- scores: preload 8 K fragments, then 8-WMMA burst ----
      v16bf kf[8];
#pragma unroll
      for (int t = 0; t < 2; ++t) {
        const bf16* kp = &Ks[buf][t * 16 + row16][hi * 8];
#pragma unroll
        for (int s = 0; s < 4; ++s)
          kf[t * 4 + s] = ld2(kp + 32 * s, kp + 32 * s + 16);
      }
      v8f st[2] = {};
#pragma unroll
      for (int s = 0; s < 4; ++s)
#pragma unroll
        for (int t = 0; t < 2; ++t)
          st[t] = WMMA_BF16(qa[s], kf[t * 4 + s], st[t]);
      // ---- causal mask (diagonal blocks only) ----
      if (j + 31 > q0) {
#pragma unroll
        for (int t = 0; t < 2; ++t) {
          int kvcol = j + t * 16 + row16;
#pragma unroll
          for (int r = 0; r < 8; ++r) {
            int qrow = q0 + r + hi * 8;
            if (kvcol > qrow) st[t][r] = -1e30f;
          }
        }
      }
      // ---- online softmax (rows split across 16-lane halves) ----
      float alpha[8];
#pragma unroll
      for (int r = 0; r < 8; ++r) {
        float mv = fmaxf(st[0][r], st[1][r]);
        mv = fmaxf(mv, __shfl_xor(mv, 1, 16));
        mv = fmaxf(mv, __shfl_xor(mv, 2, 16));
        mv = fmaxf(mv, __shfl_xor(mv, 4, 16));
        mv = fmaxf(mv, __shfl_xor(mv, 8, 16));
        float mnew = fmaxf(mrow[r], mv);
        alpha[r] = __builtin_exp2f((mrow[r] - mnew) * kScaleLog2);
        float p0 = __builtin_exp2f((st[0][r] - mnew) * kScaleLog2);
        float p1 = __builtin_exp2f((st[1][r] - mnew) * kScaleLog2);
        st[0][r] = p0;
        st[1][r] = p1;
        float rs = p0 + p1;
        rs += __shfl_xor(rs, 1, 16);
        rs += __shfl_xor(rs, 2, 16);
        rs += __shfl_xor(rs, 4, 16);
        rs += __shfl_xor(rs, 8, 16);
        lrow[r] = lrow[r] * alpha[r] + rs;
        mrow[r] = mnew;
      }
      // ---- P: C-layout -> LDS -> A-layout (bf16) ----
#pragma unroll
      for (int r = 0; r < 8; ++r) {
        plds[wave][r + hi * 8][row16]      = (bf16)st[0][r];
        plds[wave][r + hi * 8][row16 + 16] = (bf16)st[1][r];
      }
      // ---- rescale O accumulators ----
#pragma unroll
      for (int nt = 0; nt < 8; ++nt)
#pragma unroll
        for (int r = 0; r < 8; ++r)
          o[nt][r] = o[nt][r] * alpha[r];
      const bf16* pp = &plds[wave][row16][hi * 8];
      v16bf pf = ld2(pp, pp + 16);
      // ---- O += P @ V: preload 8 V fragments, then 8-WMMA burst ----
      v16bf vf[8];
#pragma unroll
      for (int nt = 0; nt < 8; ++nt) {
        const bf16* vp = &Vs[buf][nt * 16 + row16][hi * 8];
        vf[nt] = ld2(vp, vp + 16);
      }
#pragma unroll
      for (int nt = 0; nt < 8; ++nt)
        o[nt] = WMMA_BF16(pf, vf[nt], o[nt]);
    }
    __syncthreads();
    buf ^= 1;
  }

  // ---- normalize + store bf16 [B,T,H*HD] ----
  float inv[8];
#pragma unroll
  for (int r = 0; r < 8; ++r) inv[r] = 1.0f / lrow[r];
#pragma unroll
  for (int nt = 0; nt < 8; ++nt) {
#pragma unroll
    for (int r = 0; r < 8; ++r) {
      int q = q0 + r + hi * 8;
      int c = nt * 16 + row16;
      Ob[((size_t)(b * kT + q)) * (kH * kHD) + h * kHD + c] = (bf16)(o[nt][r] * inv[r]);
    }
  }
}

// ---------------------------------------------------------------------------
// Host-side orchestration
// ---------------------------------------------------------------------------
extern "C" void kernel_launch(void* const* d_in, const int* in_sizes, int n_in,
                              void* d_out, int out_size, void* d_ws, size_t ws_size,
                              hipStream_t stream) {
  const float* x  = (const float*)d_in[0];
  const float* fc = (const float*)d_in[1];
  const float* fs = (const float*)d_in[2];
  const float* Wq = (const float*)d_in[3];
  const float* Wk = (const float*)d_in[4];
  const float* Wv = (const float*)d_in[5];
  const float* Wo = (const float*)d_in[6];

  float* y  = (float*)d_out;                 // [B,T,D]           = 8388608
  float* pk = y + 8388608;                   // [B,KV,T,HD]       = 2097152
  float* pv = y + 10485760;                  // [B,KV,T,HD]       = 2097152

  char*  ws  = (char*)d_ws;
  size_t off = 0;
  auto alloc = [&](size_t bytes) {
    void* p = ws + off;
    off += (bytes + 255) & ~(size_t)255;
    return p;
  };
  bf16*  xb  = (bf16*)alloc(4096ull * 2048 * 2);   // x in bf16
  bf16*  WqT = (bf16*)alloc(2048ull * 2048 * 2);   // Wq^T
  bf16*  WkT = (bf16*)alloc( 512ull * 2048 * 2);   // Wk^T
  bf16*  WvT = (bf16*)alloc( 512ull * 2048 * 2);   // Wv^T
  bf16*  WoT = (bf16*)alloc(2048ull * 2048 * 2);   // Wo^T
  float* Qf  = (float*)alloc(4096ull * 2048 * 4);  // pre-RoPE Q
  float* Kf  = (float*)alloc(4096ull *  512 * 4);  // pre-RoPE K
  float* Vf  = (float*)alloc(4096ull *  512 * 4);  // raw V
  bf16*  Qb  = (bf16*)alloc(4096ull * 2048 * 2);   // [B,H,T,HD]
  bf16*  Kb  = (bf16*)alloc(2097152ull * 2);       // [B,KV,T,HD]
  bf16*  VbT = (bf16*)alloc(2097152ull * 2);       // [B,KV,HD,T]
  bf16*  Ob  = (bf16*)alloc(4096ull * 2048 * 2);   // attn out [B,T,H*HD]

  // precision conversion + weight transposes
  cvt_bf16_kernel<<<32768, 256, 0, stream>>>(x, xb, 8388608);
  transpose_cvt_kernel<<<16384, 256, 0, stream>>>(Wq, WqT, 2048, 2048);
  transpose_cvt_kernel<<< 4096, 256, 0, stream>>>(Wk, WkT, 2048, 512);
  transpose_cvt_kernel<<< 4096, 256, 0, stream>>>(Wv, WvT, 2048, 512);
  transpose_cvt_kernel<<<16384, 256, 0, stream>>>(Wo, WoT, 2048, 2048);

  // QKV projections (WMMA, LDS double-buffered)
  gemm_bf16<<<dim3(16, 32), 256, 0, stream>>>(xb, WqT, Qf, 4096, 2048, 2048);
  gemm_bf16<<<dim3( 4, 32), 256, 0, stream>>>(xb, WkT, Kf, 4096,  512, 2048);
  gemm_bf16<<<dim3( 4, 32), 256, 0, stream>>>(xb, WvT, Vf, 4096,  512, 2048);

  // RoPE + layout shuffles + present_k/present_v outputs
  rope_q_kernel<<<16384, 256, 0, stream>>>(Qf, fc, fs, Qb);
  rope_k_kernel<<< 4096, 256, 0, stream>>>(Kf, fc, fs, Kb, pk);
  v_prep_kernel<<< 8192, 256, 0, stream>>>(Vf, VbT, pv);

  // causal flash attention (WMMA, shared K/V tiles in LDS)
  flash_kernel<<<1024, 128, 0, stream>>>(Qb, Kb, VbT, Ob);

  // output projection (WMMA, LDS double-buffered)
  gemm_bf16<<<dim3(16, 32), 256, 0, stream>>>(Ob, WoT, y, 4096, 2048, 2048);
}
